// DPAC_43018392437116
// MI455X (gfx1250) — compile-verified, matrix operand order checked by reading, not run on hardware
//
#include <hip/hip_runtime.h>
#include <math.h>

// Problem constants (from reference)
#define BB   32
#define CC   384
#define HH   56
#define WW   56
#define HWN  (HH*WW)       // 3136
#define KTAP 7
#define KK2  49
#define EPSV 1e-6f

typedef __attribute__((ext_vector_type(2)))  float    fx2;
typedef __attribute__((ext_vector_type(8)))  float    v8f;
typedef __attribute__((ext_vector_type(16))) _Float16 v16h;
typedef int v4i __attribute__((vector_size(16)));   // matches builtin param type

// Packed fp32 FMA: 2 FMAs per lane per instruction (VOP3P v_pk_fma_f32).
__device__ __forceinline__ fx2 pk_fma(fx2 a, fx2 b, fx2 c) {
#if defined(__HIP_DEVICE_COMPILE__)
    fx2 d;
    asm("v_pk_fma_f32 %0, %1, %2, %3" : "=v"(d) : "v"(a), "v"(b), "v"(c));
    return d;
#else
    return a * b + c;
#endif
}

// Async global->LDS 16B copy (gfx1250 GLOBAL_LOAD_ASYNC_TO_LDS_B128, ASYNCcnt).
__device__ __forceinline__ void async_cp16(const float* gsrc, float* ldst) {
#if defined(__HIP_DEVICE_COMPILE__) && __has_builtin(__builtin_amdgcn_global_load_async_to_lds_b128)
    __builtin_amdgcn_global_load_async_to_lds_b128(
        (__attribute__((address_space(1))) v4i*)(void*)gsrc,
        (__attribute__((address_space(3))) v4i*)(void*)ldst,
        /*imm offset*/0, /*cpol*/0);
#else
    *(float4*)ldst = *(const float4*)gsrc;    // fallback: synchronous via VGPRs
#endif
}

__device__ __forceinline__ void wait_async0() {
#if defined(__HIP_DEVICE_COMPILE__)
#if __has_builtin(__builtin_amdgcn_s_wait_asynccnt)
    __builtin_amdgcn_s_wait_asynccnt(0);
#else
    asm volatile("s_wait_asynccnt 0x0" ::: "memory");
#endif
#endif
}

// ---------------------------------------------------------------------------
// Kernel 1: dual pooling. One block per (b,c) plane; float4 loads; LDS tree.
// Pure bandwidth (154 MB read) and warms L2 for the conv pass (x fits in 192MB L2).
// ---------------------------------------------------------------------------
__global__ __launch_bounds__(256) void k_pool(const float* __restrict__ x,
                                              float* __restrict__ pavg,
                                              float* __restrict__ pmax) {
    const int plane = blockIdx.x;                 // b*C + c
    const float4* p4 = (const float4*)(x + (size_t)plane * HWN);
    const int tid = threadIdx.x;
    float s = 0.f, m = -3.4028235e38f;
    for (int i = tid; i < HWN / 4; i += 256) {
        float4 v = p4[i];
        s += (v.x + v.y) + (v.z + v.w);
        m = fmaxf(m, fmaxf(fmaxf(v.x, v.y), fmaxf(v.z, v.w)));
    }
    __shared__ float rs[256], rm[256];
    rs[tid] = s; rm[tid] = m;
    __syncthreads();
    for (int off = 128; off > 0; off >>= 1) {
        if (tid < off) {
            rs[tid] += rs[tid + off];
            rm[tid] = fmaxf(rm[tid], rm[tid + off]);
        }
        __syncthreads();
    }
    if (tid == 0) {
        pavg[plane] = rs[0] * (1.0f / HWN);
        pmax[plane] = rm[0];
    }
}

// ---------------------------------------------------------------------------
// Kernel 2: Y = gelu(Xavg @ Wavg^T + Xmax @ Wmax^T + biases)  via fp32 WMMA.
// One wave per 16x16 tile of Y[32,384]: M tiles = 2, N tiles = 24 -> 48 waves.
// A-frag (16x4 f32): lanes 0-15 = M rows, elem0/elem1 hold K=kb/kb+1,
// lanes 16-31 give kb=2. B-frag mirrors it with N on the lanes.
// ---------------------------------------------------------------------------
__global__ __launch_bounds__(32) void k_mm(const float* __restrict__ pavg,
                                           const float* __restrict__ pmax,
                                           const float* __restrict__ w_avg,
                                           const float* __restrict__ b_avg,
                                           const float* __restrict__ w_max,
                                           const float* __restrict__ b_max,
                                           float* __restrict__ yact) {
    const int lane = threadIdx.x;          // 0..31, EXEC all-ones (WMMA req.)
    const int mt   = blockIdx.x & 1;       // 2 M-tiles
    const int nt   = blockIdx.x >> 1;      // 24 N-tiles
    const int nl   = lane & 15;
    const int half = lane >> 4;
    const int nG   = nt * 16 + nl;         // output channel
    const int mG   = mt * 16 + nl;         // batch row (A fragment rows)

    const float* arow_avg = pavg + (size_t)mG * CC;
    const float* arow_max = pmax + (size_t)mG * CC;
    const float* brow_avg = w_avg + (size_t)nG * CC;  // B[k,n] = w[n,k]
    const float* brow_max = w_max + (size_t)nG * CC;

    v8f acc = {};

#if defined(__HIP_DEVICE_COMPILE__) && __has_builtin(__builtin_amdgcn_wmma_f32_16x16x4_f32)
    const int kb = half * 2;               // lanes 16-31 carry K=2,3
    for (int k = 0; k < CC; k += 4) {
        fx2 aa = *(const fx2*)(arow_avg + k + kb);
        fx2 ba = *(const fx2*)(brow_avg + k + kb);
        acc = __builtin_amdgcn_wmma_f32_16x16x4_f32(false, aa, false, ba,
                                                    (short)0, acc, false, false);
        fx2 am = *(const fx2*)(arow_max + k + kb);
        fx2 bm = *(const fx2*)(brow_max + k + kb);
        acc = __builtin_amdgcn_wmma_f32_16x16x4_f32(false, am, false, bm,
                                                    (short)0, acc, false, false);
    }
#elif defined(__HIP_DEVICE_COMPILE__) && __has_builtin(__builtin_amdgcn_wmma_f32_16x16x32_f16)
    // Fallback: codegen-confirmed f16 WMMA, f32 accumulate.
    for (int k0 = 0; k0 < CC; k0 += 32) {
        v16h a, b, a2, b2;
        #pragma unroll
        for (int e = 0; e < 16; ++e) {
            int kk = k0 + half * 8 + e + (e < 8 ? 0 : 8);
            a[e]  = (_Float16)arow_avg[kk];
            b[e]  = (_Float16)brow_avg[kk];
            a2[e] = (_Float16)arow_max[kk];
            b2[e] = (_Float16)brow_max[kk];
        }
        acc = __builtin_amdgcn_wmma_f32_16x16x32_f16(false, a, false, b,
                                                     (short)0, acc, false, false);
        acc = __builtin_amdgcn_wmma_f32_16x16x32_f16(false, a2, false, b2,
                                                     (short)0, acc, false, false);
    }
#else
    // Host-pass stub (never executed on device).
    for (int k = 0; k < CC; ++k)
        acc[0] += arow_avg[k] * brow_avg[k] + arow_max[k] * brow_max[k];
#endif

    // D layout: VGPR r -> M = r (lanes 0-15) / r+8 (lanes 16-31), N = lane&15.
    const float bias = b_avg[nG] + b_max[nG];
    #pragma unroll
    for (int r = 0; r < 8; ++r) {
        int mo = mt * 16 + r + half * 8;
        float v = acc[r] + bias;
        v = 0.5f * v * (1.0f + erff(v * 0.70710678118654752f));  // exact GELU
        yact[(size_t)mo * CC + nG] = v;
    }
}

// ---------------------------------------------------------------------------
// Kernel 3: FGRN + per-sample kernel build.
// Gx[b,c] = |y[b,c]| * ||w_mix[c,:]||  (the 49-wide reduction factors out).
// ---------------------------------------------------------------------------
__global__ __launch_bounds__(384) void k_fgrn(const float* __restrict__ yact,
                                              const float* __restrict__ w_mix,
                                              const float* __restrict__ gamma,
                                              const float* __restrict__ beta,
                                              float* __restrict__ kern) {
    const int b = blockIdx.x, c = threadIdx.x;
    const float y = yact[(size_t)b * CC + c];
    const float* wm = w_mix + (size_t)c * KK2;
    float wn2 = 0.f;
    #pragma unroll
    for (int j = 0; j < KK2; ++j) { float w = wm[j]; wn2 = fmaf(w, w, wn2); }
    const float Gx = fabsf(y) * sqrtf(wn2);

    __shared__ float red[512];
    red[c] = Gx;
    if (c < 128) red[c + 384] = 0.f;
    __syncthreads();
    for (int off = 256; off > 0; off >>= 1) {
        if (c < off) red[c] += red[c + off];
        __syncthreads();
    }
    const float meanG = red[0] * (1.0f / CC);
    const float Nx = Gx / (meanG + EPSV);
    const float s  = gamma[c] * y * Nx;
    const float bt = beta[c];
    float* kd = kern + ((size_t)b * CC + c) * KK2;
    #pragma unroll
    for (int j = 0; j < KK2; ++j) kd[j] = fmaf(s, wm[j], bt);
}

// ---------------------------------------------------------------------------
// Kernel 4: per-sample 7x7 depthwise conv. One block per (b,c) plane.
// 56x56 tile async-DMA'd into LDS (GLOBAL_LOAD_ASYNC_TO_LDS_B128, ASYNCcnt),
// zeroed halo: rows +3, cols +4 (16B-aligned interior for B128), stride 64.
// Compute: float2 outputs via v_pk_fma_f32 (2 FMA/lane/instr) -> fp32 rate
// doubles, pushing the 3.78 GFLOP stage toward the 23.3 TB/s memory roof;
// the second read of x should hit the 192 MB L2 warmed by k_pool.
// ---------------------------------------------------------------------------
#define LDSW 64
__global__ __launch_bounds__(256) void k_dwconv(const float* __restrict__ x,
                                                const float* __restrict__ kern,
                                                float* __restrict__ out) {
    const int plane = blockIdx.x;
    const float* xp = x + (size_t)plane * HWN;
    float* op = out + (size_t)plane * HWN;
    const int tid = threadIdx.x;

    __shared__ __align__(16) float tile[62 * LDSW];
    __shared__ float kw[KK2];

    // Zero the whole tile (halo included) with aligned float4 LDS stores.
    for (int i = tid; i < (62 * LDSW) / 4; i += 256) ((float4*)tile)[i] = make_float4(0.f, 0.f, 0.f, 0.f);
    if (tid < KK2) kw[tid] = kern[(size_t)plane * KK2 + tid];
    __syncthreads();                         // zeros visible before async writes

    // Interior: x row r, col c  ->  tile[(r+3)*64 + (c+4)].
    // 784 float4 chunks; c multiple of 4 -> LDS dest (c+4) is 16B-aligned.
    for (int g = tid; g < HWN / 4; g += 256) {
        int r  = g / (WW / 4);
        int c  = (g - r * (WW / 4)) * 4;
        async_cp16(xp + r * WW + c, &tile[(r + 3) * LDSW + (c + 4)]);
    }
    wait_async0();                           // drain this wave's ASYNCcnt
    __syncthreads();                         // all waves' tiles landed

    // out[h][w] needs x[h+i-3][w+j-3] -> tile row (h+i), col (w+j+1).
    for (int u = tid; u < HH * (WW / 2); u += 256) {      // 1568 float2 outputs
        const int h  = u / (WW / 2);
        const int w2 = (u - h * (WW / 2)) * 2;
        fx2 acc = {0.f, 0.f};
        #pragma unroll
        for (int i = 0; i < KTAP; ++i) {
            const float* row = &tile[(h + i) * LDSW + w2 + 1];
            float xr[8];
            #pragma unroll
            for (int t = 0; t < 8; ++t) xr[t] = row[t];   // taps overlap: 8 loads/row
            #pragma unroll
            for (int j = 0; j < KTAP; ++j) {
                float wv = kw[i * KTAP + j];
                fx2 wvv = {wv, wv};
                fx2 xv  = {xr[j], xr[j + 1]};
                acc = pk_fma(wvv, xv, acc);
            }
        }
        *(fx2*)(op + h * WW + w2) = acc;                  // 8B-aligned (w2 even)
    }
}

// ---------------------------------------------------------------------------
extern "C" void kernel_launch(void* const* d_in, const int* in_sizes, int n_in,
                              void* d_out, int out_size, void* d_ws, size_t ws_size,
                              hipStream_t stream) {
    const float* x     = (const float*)d_in[0];
    const float* w_avg = (const float*)d_in[1];
    const float* b_avg = (const float*)d_in[2];
    const float* w_max = (const float*)d_in[3];
    const float* b_max = (const float*)d_in[4];
    const float* w_mix = (const float*)d_in[5];
    const float* gamma = (const float*)d_in[6];
    const float* beta  = (const float*)d_in[7];
    float* out = (float*)d_out;

    // Workspace layout (floats): pavg[BC] | pmax[BC] | yact[BC] | kern[BC*49]
    float* ws   = (float*)d_ws;
    float* pavg = ws;
    float* pmax = ws + (size_t)BB * CC;
    float* yact = ws + (size_t)2 * BB * CC;
    float* kern = ws + (size_t)3 * BB * CC;   // total ~2.56 MB

    k_pool  <<<BB * CC, 256, 0, stream>>>(x, pavg, pmax);
    k_mm    <<<48,      32,  0, stream>>>(pavg, pmax, w_avg, b_avg, w_max, b_max, yact);
    k_fgrn  <<<BB,      384, 0, stream>>>(yact, w_mix, gamma, beta, kern);
    k_dwconv<<<BB * CC, 256, 0, stream>>>(x, kern, out);
}
